// gruaging_rnnosc3ode_70308614636171
// MI455X (gfx1250) — matrix operand order repaired and checked by reading, not compiled, not run on hardware
//
#include <hip/hip_runtime.h>
#include <hip/hip_bf16.h>
#include <cstddef>

// ---------------- problem constants (from the reference) ----------------
constexpr int B_ = 256;
constexpr int T_ = 2048;
constexpr int NIN = 8;          // NINPUT
constexpr int NXC = 9;          // NINPUT + 1 (GRU input width)
constexpr int NHA = 128;        // NH_AGE
constexpr int NG  = 3 * NHA;    // 384 gate columns
constexpr int NHI = 64;         // NH_IN
constexpr int NHO = 64;         // NH_OUT
constexpr float W0C = -20.0f;
constexpr float W1C = 10.0f;
constexpr float DT  = 0.02f;    // HRNN * 1e9

// ---------------- vector types for WMMA ----------------
typedef __attribute__((ext_vector_type(16))) __bf16 v16bf;
typedef __attribute__((ext_vector_type(8)))  __bf16 v8bf;
typedef __attribute__((ext_vector_type(8)))  float  v8f;

// fast transcendental helpers: v_exp_f32 + v_rcp_f32, no IEEE-div expansion
__device__ __forceinline__ float sigf(float x) {
    return __builtin_amdgcn_rcpf(1.0f + __expf(-x));
}
__device__ __forceinline__ float tanh_fast(float x) {
    return 1.0f - 2.0f * __builtin_amdgcn_rcpf(__expf(2.0f * x) + 1.0f);
}

// D = A(16xK, batch rows) x B(Kx16, weights) + C, bf16 in / f32 acc
__device__ __forceinline__ v8f wmma_bf16(v16bf a, v16bf b, v8f c) {
    return __builtin_amdgcn_wmma_f32_16x16x32_bf16(
        false, a, false, b, (short)0, c, false, false);
}

// 16-bit A-matrix 16x32 tile, contiguous k order (lane half h holds k = 16h + s), same as load_b16
__device__ __forceinline__ v16bf load_a16(const __bf16* base, int ldk, int lane, int k0) {
    int m    = lane & 15;
    int koff = (lane >> 4) << 4;                 // 0 or 16
    const __bf16* p = base + m * ldk + k0 + koff;
    v8bf lo = *(const v8bf*)(p);
    v8bf hi = *(const v8bf*)(p + 8);
    return __builtin_shufflevector(lo, hi, 0,1,2,3,4,5,6,7,8,9,10,11,12,13,14,15);
}

// 16-bit B-matrix 32x16 tile; weights stored LDS-[n][k] -> contiguous 32B per lane
__device__ __forceinline__ v16bf load_b16(const __bf16* base, int ldk, int lane,
                                          int ncol0, int k0) {
    int n  = ncol0 + (lane & 15);
    int kb = k0 + ((lane >> 4) << 4);            // 0 or 16
    return *(const v16bf*)(base + n * ldk + kb);
}

// -------- async global->LDS (ASYNCcnt-tracked), inline asm (no builtin) -----
__device__ __forceinline__ unsigned lds_off_of(const void* p) {
    // generic shared address: low 32 bits are the LDS offset (aperture in MSBs)
    return (unsigned)(unsigned long long)p;
}
__device__ __forceinline__ void async_b128(unsigned lds_off, const float* g) {
    asm volatile("global_load_async_to_lds_b128 %0, %1, off"
                 :: "v"(lds_off), "v"(g) : "memory");
}
__device__ __forceinline__ void wait_async0()  { asm volatile("s_wait_asynccnt 0"  ::: "memory"); }
__device__ __forceinline__ void wait_async16() { asm volatile("s_wait_asynccnt 16" ::: "memory"); }

// ======================================================================
// Kernel 1: GRU scan.  grid = B/16 blocks, 256 threads (8 waves).
// Wave w owns gate columns [16w,16w+16) of r/z/n; h register-resident.
// Weights LDS-resident once; B-operand tiles + biases hoisted to registers;
// h and x double-buffered -> ONE s_barrier per timestep.
// ======================================================================
__global__ __launch_bounds__(256, 1)
void gru_scan_kernel(const float* __restrict__ x,
                     const float* __restrict__ Wih, const float* __restrict__ Whh,
                     const float* __restrict__ bih, const float* __restrict__ bhh,
                     float* __restrict__ ws_hT)
{
    extern __shared__ char smem[];
    __bf16* lds_whh = (__bf16*)smem;              // 384*128
    __bf16* lds_wih = lds_whh + NG * NHA;         // 384*32
    __bf16* lds_h   = lds_wih + NG * 32;          // 2 * 16*128 (double buffer)
    __bf16* lds_x   = lds_h   + 2 * 16 * NHA;     // 2 * 16*32  (double buffer)
    float*  lds_bih = (float*)(lds_x + 2 * 16 * 32);
    float*  lds_bhh = lds_bih + NG;

    const int tid  = threadIdx.x;
    const int lane = tid & 31;
    const int w    = tid >> 5;                    // wave 0..7
    const int b0   = blockIdx.x * 16;             // batch tile base

    // ---- stage weights once (f32 -> bf16) ----
    for (int i = tid; i < NG * NHA; i += 256) lds_whh[i] = (__bf16)Whh[i];
    for (int i = tid; i < NG * 32;  i += 256) {
        int n = i >> 5, k = i & 31;
        lds_wih[i] = (k < NXC) ? (__bf16)Wih[n * NXC + k] : (__bf16)0.0f;
    }
    for (int i = tid; i < NG; i += 256) { lds_bih[i] = bih[i]; lds_bhh[i] = bhh[i]; }
    for (int i = tid; i < 2 * 16 * NHA; i += 256) lds_h[i] = (__bf16)0.0f;
    for (int i = tid; i < 2 * 16 * 32;  i += 256) lds_x[i] = (__bf16)0.0f;
    __syncthreads();

    // ---- hoist loop-invariant B tiles and biases into registers ----
    const int ncol  = lane & 15;
    const int col_r = 0 * NHA + w * 16 + ncol;
    const int col_z = 1 * NHA + w * 16 + ncol;
    const int col_n = 2 * NHA + w * 16 + ncol;
    const float bir = lds_bih[col_r] + lds_bhh[col_r];
    const float biz = lds_bih[col_z] + lds_bhh[col_z];
    const float bin = lds_bih[col_n];
    const float bhn = lds_bhh[col_n];

    v16bf Bxr = load_b16(lds_wih, 32, lane, 0 * NHA + w * 16, 0);
    v16bf Bxz = load_b16(lds_wih, 32, lane, 1 * NHA + w * 16, 0);
    v16bf Bxn = load_b16(lds_wih, 32, lane, 2 * NHA + w * 16, 0);
    v16bf Br[4], Bz[4], Bn[4];
    #pragma unroll
    for (int c = 0; c < 4; ++c) {
        Br[c] = load_b16(lds_whh, NHA, lane, 0 * NHA + w * 16, 32 * c);
        Bz[c] = load_b16(lds_whh, NHA, lane, 1 * NHA + w * 16, 32 * c);
        Bn[c] = load_b16(lds_whh, NHA, lane, 2 * NHA + w * 16, 32 * c);
    }

    // stage x_0 into buffer 0
    if (tid < 16 * NXC) {
        int m = tid / NXC, c = tid % NXC;
        lds_x[m * 32 + c] = (__bf16)x[((size_t)(b0 + m) * T_ + 0) * NXC + c];
    }
    __syncthreads();

    v8f h;
    #pragma unroll
    for (int i = 0; i < 8; ++i) h[i] = 0.0f;

    for (int t = 0; t < T_; ++t) {
        const __bf16* hbuf = lds_h + (t & 1) * 16 * NHA;
        const __bf16* xbuf = lds_x + (t & 1) * 16 * 32;
        __bf16* hnxt = lds_h + ((t + 1) & 1) * 16 * NHA;
        __bf16* xnxt = lds_x + ((t + 1) & 1) * 16 * 32;

        // register-stage x_{t+1}: global latency hides under the WMMA chain
        float xreg = 0.0f; int xm = 0, xc = 0;
        if (tid < 16 * NXC && t + 1 < T_) {
            xm = tid / NXC; xc = tid % NXC;
            xreg = x[((size_t)(b0 + xm) * T_ + (t + 1)) * NXC + xc];
        }
        if (lane == 0 && t + 8 < T_)
            __builtin_prefetch(&x[((size_t)(b0 + w * 2) * T_ + (t + 8)) * NXC], 0, 1);

        v16bf ax  = load_a16(xbuf, 32,  lane, 0);
        v16bf ah0 = load_a16(hbuf, NHA, lane, 0);
        v16bf ah1 = load_a16(hbuf, NHA, lane, 32);
        v16bf ah2 = load_a16(hbuf, NHA, lane, 64);
        v16bf ah3 = load_a16(hbuf, NHA, lane, 96);

        v8f ar, az, ain, ahn;
        #pragma unroll
        for (int i = 0; i < 8; ++i) { ar[i]=bir; az[i]=biz; ain[i]=bin; ahn[i]=bhn; }

        ar  = wmma_bf16(ax,  Bxr,   ar);
        az  = wmma_bf16(ax,  Bxz,   az);
        ain = wmma_bf16(ax,  Bxn,   ain);
        ar  = wmma_bf16(ah0, Br[0], ar);
        ar  = wmma_bf16(ah1, Br[1], ar);
        ar  = wmma_bf16(ah2, Br[2], ar);
        ar  = wmma_bf16(ah3, Br[3], ar);
        az  = wmma_bf16(ah0, Bz[0], az);
        az  = wmma_bf16(ah1, Bz[1], az);
        az  = wmma_bf16(ah2, Bz[2], az);
        az  = wmma_bf16(ah3, Bz[3], az);
        ahn = wmma_bf16(ah0, Bn[0], ahn);
        ahn = wmma_bf16(ah1, Bn[1], ahn);
        ahn = wmma_bf16(ah2, Bn[2], ahn);
        ahn = wmma_bf16(ah3, Bn[3], ahn);

        // gate combine, entirely in C-layout registers (fully unrolled)
        #pragma unroll
        for (int i = 0; i < 8; ++i) {
            float r = sigf(ar[i]);
            float z = sigf(az[i]);
            float n = tanh_fast(ain[i] + r * ahn[i]);
            h[i] = (1.0f - z) * n + z * h[i];
        }

        {   // write new h (bf16) into the other buffer
            int m0 = (lane >> 4) * 8;
            int hc = w * 16 + ncol;
            #pragma unroll
            for (int i = 0; i < 8; ++i) hnxt[(m0 + i) * NHA + hc] = (__bf16)h[i];
        }
        if (tid < 16 * NXC && t + 1 < T_) xnxt[xm * 32 + xc] = (__bf16)xreg;
        __syncthreads();                 // single barrier per step (double buffered)
    }

    int m0 = (lane >> 4) * 8;
    int hc = w * 16 + ncol;
    #pragma unroll
    for (int i = 0; i < 8; ++i)
        ws_hT[(size_t)(b0 + m0 + i) * NHA + hc] = h[i];
}

// ======================================================================
// Kernel 2: dparams[b] = hT[b] . W_age + b_age
// ======================================================================
__global__ void dparams_kernel(const float* __restrict__ hT,
                               const float* __restrict__ W_age,
                               const float* __restrict__ b_age,
                               float* __restrict__ dp)
{
    int b = blockIdx.x * blockDim.x + threadIdx.x;
    if (b < B_) {
        float s = b_age[0];
        for (int k = 0; k < NHA; ++k) s += hT[(size_t)b * NHA + k] * W_age[k];
        dp[b] = s;
    }
}

// ======================================================================
// Kernel 3: inv_scale at nodes and midpoints.  One thread per (b,t).
// ======================================================================
__global__ __launch_bounds__(256)
void scale_kernel(const float* __restrict__ x,
                  const float* __restrict__ Wnin, const float* __restrict__ bnin,
                  const float* __restrict__ Wlin, const float* __restrict__ blin,
                  const float* __restrict__ tau,  const float* __restrict__ dp,
                  float* __restrict__ inv_s, float* __restrict__ inv_m)
{
    __shared__ float sw[NHI * NIN];
    __shared__ float sb[NHI];
    __shared__ float sl[NHI];
    for (int i = threadIdx.x; i < NHI * NIN; i += 256) sw[i] = Wnin[i];
    if (threadIdx.x < NHI) { sb[threadIdx.x] = bnin[threadIdx.x]; sl[threadIdx.x] = Wlin[threadIdx.x]; }
    __syncthreads();

    size_t g = (size_t)blockIdx.x * 256 + threadIdx.x;
    if (g >= (size_t)B_ * T_) return;
    int b = (int)(g / T_);
    int t = (int)(g % T_);

    const float* p = &x[g * NXC];
    float u[NIN], um[NIN];
    #pragma unroll
    for (int c = 0; c < NIN; ++c) u[c] = p[c];
    bool hasmid = (t < T_ - 1);
    if (hasmid) {
        #pragma unroll
        for (int c = 0; c < NIN; ++c) um[c] = 0.5f * (u[c] + p[NXC + c]);
    }

    float base = tau[0] + dp[b] + blin[0];
    float alo = 0.0f, amid = 0.0f;
    for (int j = 0; j < NHI; ++j) {
        float s1 = sb[j], s2 = sb[j];
        #pragma unroll
        for (int c = 0; c < NIN; ++c) {
            float wv = sw[j * NIN + c];
            s1 += wv * u[c];
            if (hasmid) s2 += wv * um[c];
        }
        alo += sigf(s1) * sl[j];
        if (hasmid) amid += sigf(s2) * sl[j];
    }
    inv_s[g] = __expf(-(base + alo));
    if (hasmid) inv_m[g] = __expf(-(base + amid));
}

// ======================================================================
// Kernel 4: sequential RK4 scan.  One wave (32 threads) per 32 batch rows.
// inv_s/inv_m staged in 32-timestep chunks via global_load_async_to_lds_b128
// (coalesced 16B/lane), double-buffered: next chunk's async DMA overlaps the
// tanh-latency-bound RK4 math; s_wait_asynccnt 16 releases the current chunk.
// ======================================================================
constexpr int OCH = 32;                     // chunk width (timesteps)
constexpr int OPAD = 36;                    // row pitch (floats), 16B-aligned

__global__ __launch_bounds__(32)
void ode_kernel(const float* __restrict__ inv_s,
                const float* __restrict__ inv_m,
                float* __restrict__ ws_x)
{
    __shared__ __attribute__((aligned(16))) float lds_s[2][32][OPAD];
    __shared__ __attribute__((aligned(16))) float lds_m[2][32][OPAD];

    const int lane = threadIdx.x;           // 0..31
    const int b0   = blockIdx.x * 32;
    const int b    = b0 + lane;

    // issue one chunk: 8 b128 per array, 4 rows per instruction
    auto stage = [&](int buf, int t0) {
        int rsub = lane >> 3;               // 0..3
        int c    = (lane & 7) * 4;          // 0..28
        #pragma unroll
        for (int i = 0; i < 8; ++i) {
            int r = 4 * i + rsub;
            async_b128(lds_off_of(&lds_s[buf][r][c]),
                       &inv_s[(size_t)(b0 + r) * T_ + t0 + c]);
            async_b128(lds_off_of(&lds_m[buf][r][c]),
                       &inv_m[(size_t)(b0 + r) * T_ + t0 + c]);
        }
    };

    float x1 = 0.4736f, x2 = 0.8745f, x3 = 1.8497f;
    float* xr = &ws_x[(size_t)b * T_ * 3];
    xr[0] = x1; xr[1] = x2; xr[2] = x3;

    stage(0, 0);

    const int NCH = T_ / OCH;               // 64
    for (int ch = 0; ch < NCH; ++ch) {
        if (ch + 1 < NCH) { stage((ch + 1) & 1, (ch + 1) * OCH); wait_async16(); }
        else              { wait_async0(); }

        const float (*cs)[OPAD] = lds_s[ch & 1];
        const float (*cm)[OPAD] = lds_m[ch & 1];
        const int t0 = ch * OCH;

        for (int tl = 0; tl < OCH; ++tl) {
            int t = t0 + tl;
            if (t >= T_ - 1) break;
            float s1 = cs[lane][tl];
            float sm = cm[lane][tl];
            float s4 = (tl < OCH - 1) ? cs[lane][tl + 1]
                                      : inv_s[(size_t)b * T_ + t + 1];
            #define FODE(a1,a2,a3,s,d1,d2,d3)                                       \
                { float t1 = tanh_fast(a1), t2 = tanh_fast(a2), t3 = tanh_fast(a3); \
                  d1 = (-(a1) + t2) * (s);                                          \
                  d2 = (-(a2) + t3) * (s);                                          \
                  d3 = (-(a3) + W0C * t1 + W1C * t2) * (s); }
            float k11,k12,k13, k21,k22,k23, k31,k32,k33, k41,k42,k43;
            FODE(x1, x2, x3, s1, k11, k12, k13);
            FODE(x1 + 0.5f*DT*k11, x2 + 0.5f*DT*k12, x3 + 0.5f*DT*k13, sm, k21, k22, k23);
            FODE(x1 + 0.5f*DT*k21, x2 + 0.5f*DT*k22, x3 + 0.5f*DT*k23, sm, k31, k32, k33);
            FODE(x1 + DT*k31,      x2 + DT*k32,      x3 + DT*k33,      s4, k41, k42, k43);
            #undef FODE
            x1 += DT / 6.0f * (k11 + 2.0f*k21 + 2.0f*k31 + k41);
            x2 += DT / 6.0f * (k12 + 2.0f*k22 + 2.0f*k32 + k42);
            x3 += DT / 6.0f * (k13 + 2.0f*k23 + 2.0f*k33 + k43);
            xr[(t + 1) * 3 + 0] = x1;
            xr[(t + 1) * 3 + 1] = x2;
            xr[(t + 1) * 3 + 2] = x3;
        }
    }
}

// ======================================================================
// Kernel 5: output head via bf16 WMMA over 16-row tiles of (B*T) rows.
// ======================================================================
__global__ __launch_bounds__(256)
void head_kernel(const float* __restrict__ xs,
                 const float* __restrict__ Wnl, const float* __restrict__ bnl,
                 const float* __restrict__ Wout, const float* __restrict__ bout,
                 float* __restrict__ out)
{
    __shared__ __attribute__((aligned(32))) __bf16 lds_wnl[NHO * 32];
    __shared__ __attribute__((aligned(32))) __bf16 lds_a[8][16 * 32];
    __shared__ float lds_bnl[NHO];
    __shared__ float lds_wo[NHO];

    for (int i = threadIdx.x; i < NHO * 32; i += 256) {
        int n = i >> 5, k = i & 31;
        lds_wnl[i] = (k < 3) ? (__bf16)Wnl[n * 3 + k] : (__bf16)0.0f;
    }
    if (threadIdx.x < NHO) { lds_bnl[threadIdx.x] = bnl[threadIdx.x]; lds_wo[threadIdx.x] = Wout[threadIdx.x]; }
    for (int i = threadIdx.x; i < 8 * 16 * 32; i += 256) ((__bf16*)lds_a)[i] = (__bf16)0.0f;
    __syncthreads();

    const int lane = threadIdx.x & 31;
    const int w    = threadIdx.x >> 5;
    const float bo = bout[0];

    // hoist B tiles, per-lane bias/weight scalars
    v16bf Bt[4];
    float bias[4], wv[4];
    #pragma unroll
    for (int nt = 0; nt < 4; ++nt) {
        Bt[nt]   = load_b16(lds_wnl, 32, lane, nt * 16, 0);
        bias[nt] = lds_bnl[nt * 16 + (lane & 15)];
        wv[nt]   = lds_wo [nt * 16 + (lane & 15)];
    }

    const int ntiles = (B_ * T_) / 16;             // 32768
    for (int tt = blockIdx.x * 8 + w; tt < ntiles; tt += gridDim.x * 8) {
        size_t r0 = (size_t)tt * 16;
        if (lane < 16) {
            const float* p = &xs[(r0 + lane) * 3];
            __bf16* a = &lds_a[w][lane * 32];
            a[0] = (__bf16)p[0]; a[1] = (__bf16)p[1]; a[2] = (__bf16)p[2];
        }
        v16bf a = load_a16(lds_a[w], 32, lane, 0);

        v8f part;
        #pragma unroll
        for (int i = 0; i < 8; ++i) part[i] = 0.0f;
        #pragma unroll
        for (int nt = 0; nt < 4; ++nt) {
            v8f acc;
            #pragma unroll
            for (int i = 0; i < 8; ++i) acc[i] = bias[nt];
            acc = wmma_bf16(a, Bt[nt], acc);
            #pragma unroll
            for (int i = 0; i < 8; ++i) part[i] += sigf(acc[i]) * wv[nt];
        }
        #pragma unroll
        for (int mask = 1; mask < 16; mask <<= 1) {
            #pragma unroll
            for (int i = 0; i < 8; ++i) part[i] += __shfl_xor(part[i], mask);
        }
        if ((lane & 15) == 0) {
            int mrow = (lane >> 4) * 8;
            #pragma unroll
            for (int i = 0; i < 8; ++i) out[r0 + mrow + i] = part[i] + bo;
        }
    }
}

// ======================================================================
extern "C" void kernel_launch(void* const* d_in, const int* in_sizes, int n_in,
                              void* d_out, int out_size, void* d_ws, size_t ws_size,
                              hipStream_t stream)
{
    const float* inputdata = (const float*)d_in[0];
    const float* W_ih  = (const float*)d_in[1];
    const float* W_hh  = (const float*)d_in[2];
    const float* b_ih  = (const float*)d_in[3];
    const float* b_hh  = (const float*)d_in[4];
    const float* W_age = (const float*)d_in[5];
    const float* b_age = (const float*)d_in[6];
    const float* W_nin = (const float*)d_in[7];
    const float* b_nin = (const float*)d_in[8];
    const float* W_lin = (const float*)d_in[9];
    const float* b_lin = (const float*)d_in[10];
    const float* tau   = (const float*)d_in[11];
    const float* W_nl  = (const float*)d_in[12];
    const float* b_nl  = (const float*)d_in[13];
    const float* W_out = (const float*)d_in[14];
    const float* b_out = (const float*)d_in[15];
    float* out = (float*)d_out;

    // workspace partition (floats): hT | dp | inv_s | inv_mid | traj
    float* ws      = (float*)d_ws;
    float* ws_hT   = ws;                                   // 256*128
    float* ws_dp   = ws_hT + (size_t)B_ * NHA;             // 256
    float* ws_s    = ws_dp + B_;                           // B*T
    float* ws_m    = ws_s + (size_t)B_ * T_;               // B*T
    float* ws_traj = ws_m + (size_t)B_ * T_;               // B*T*3

    // GRU: whh 96K + wih 24K + 2x h 8K + 2x x 2K + biases 3K = 136192 B LDS
    constexpr size_t GRU_LDS =
        (size_t)NG * NHA * 2 + (size_t)NG * 32 * 2 +
        2 * 16 * NHA * 2 + 2 * 16 * 32 * 2 + 2 * NG * 4;
    gru_scan_kernel<<<B_ / 16, 256, GRU_LDS, stream>>>(
        inputdata, W_ih, W_hh, b_ih, b_hh, ws_hT);

    dparams_kernel<<<1, 256, 0, stream>>>(ws_hT, W_age, b_age, ws_dp);

    scale_kernel<<<(B_ * T_) / 256, 256, 0, stream>>>(
        inputdata, W_nin, b_nin, W_lin, b_lin, tau, ws_dp, ws_s, ws_m);

    ode_kernel<<<B_ / 32, 32, 0, stream>>>(ws_s, ws_m, ws_traj);

    head_kernel<<<512, 256, 0, stream>>>(ws_traj, W_nl, b_nl, W_out, b_out, out);
}
